// MultiHeadAttention_1898375545091
// MI455X (gfx1250) — matrix operand order repaired and checked by reading, not compiled, not run on hardware
//
#include <hip/hip_runtime.h>

#define B_ 2
#define S_ 2048
#define D_ 1024
#define H_ 16
#define DEPTH_ 64
#define BS_ (B_ * S_)
#define NEGINF_ (-1.0e9f)

typedef __bf16 bf16;
typedef __attribute__((ext_vector_type(16))) __bf16 v16bf;
typedef __attribute__((ext_vector_type(8)))  float  v8f;
typedef __attribute__((ext_vector_type(4)))  unsigned int u32x4;
typedef __attribute__((ext_vector_type(8)))  int i32x8;
typedef __attribute__((ext_vector_type(4)))  int i32x4;

// ---------------------------------------------------------------------------
// WMMA wrapper: D = A(16x32 bf16) * B(32x16 bf16) + C(16x16 f32)
// ---------------------------------------------------------------------------
__device__ __forceinline__ v8f wmma_bf16(v16bf a, v16bf b, v8f c) {
    return __builtin_amdgcn_wmma_f32_16x16x32_bf16(
        /*neg_a=*/false, a, /*neg_b=*/false, b,
        /*c_mod=*/(short)0, c, /*reuse_a=*/false, /*reuse_b=*/false);
}

// A-matrix fragment, 16x32 bf16 (ISA 7.12.2: lanes 0-15 row M=lane, K={0..7,16..23};
// lanes 16-31 row M=lane-16, K={8..15,24..31})
__device__ __forceinline__ v16bf load_a_bf16(const bf16* __restrict__ A, int lda,
                                             int row0, int k0, int lane) {
    const int r = lane & 15;
    const int koff = (lane < 16) ? 0 : 8;
    const bf16* p = A + (size_t)(row0 + r) * lda + k0 + koff;
    v16bf a;
#pragma unroll
    for (int i = 0; i < 8; ++i) a[i] = p[i];
#pragma unroll
    for (int i = 0; i < 8; ++i) a[8 + i] = p[16 + i];
    return a;
}

__device__ __forceinline__ v16bf load_a_f32(const float* __restrict__ A, int lda,
                                            int row0, int k0, int lane) {
    const int r = lane & 15;
    const int koff = (lane < 16) ? 0 : 8;
    const float* p = A + (size_t)(row0 + r) * lda + k0 + koff;
    v16bf a;
#pragma unroll
    for (int i = 0; i < 8; ++i) a[i] = (bf16)p[i];
#pragma unroll
    for (int i = 0; i < 8; ++i) a[8 + i] = (bf16)p[16 + i];
    return a;
}

// B-matrix fragment, 32x16 bf16 (lanes 0-15: col N=lane, K=k0+{0..7,16..23};
// lanes 16-31: col N=lane-16, K=k0+{8..15,24..31}). Bt column-major: Bt[n*ldb + k].
// Works for both global and LDS source pointers.
__device__ __forceinline__ v16bf load_b_bf16(const bf16* Bt, int ldb,
                                             int col0, int k0, int lane) {
    const int n = col0 + (lane & 15);
    const int koff = (lane < 16) ? 0 : 8;
    const bf16* p = Bt + (size_t)n * ldb + k0 + koff;
    v16bf b;
#pragma unroll
    for (int i = 0; i < 8; ++i) b[i] = p[i];
#pragma unroll
    for (int i = 0; i < 8; ++i) b[8 + i] = p[16 + i];
    return b;
}

// ---------------------------------------------------------------------------
// TDM: async 2D tile load (rows x tile_k bf16) from global into LDS.
// Descriptor per ISA ch.8: group0 = {count, lds_addr, global_addr, type=2},
// group1 = {data_size=2B, tensor dims, tile dims, strides}.
// 6-arg toolchain form: (g0, g1, g2, g3, g4, cpol); g2..g4 unused for 2D tiles.
// ---------------------------------------------------------------------------
__device__ __forceinline__ void tdm_load_2d(unsigned lds_addr, const bf16* gptr,
                                            int tensor_k, int tensor_rows,
                                            int tile_k, int tile_rows, int stride_k) {
    unsigned long long ga = (unsigned long long)(size_t)gptr;
    u32x4 g0;
    g0[0] = 1u;                                        // count=1, user descriptor
    g0[1] = lds_addr;                                  // LDS byte address
    g0[2] = (unsigned)(ga & 0xffffffffull);            // global_addr[31:0]
    g0[3] = (unsigned)((ga >> 32) & 0x01ffffffull) | (2u << 30);  // addr[56:32] | type=2
    i32x8 g1;
    g1[0] = (int)(1u << 16);                           // data_size=1 -> 2 bytes
    g1[1] = (int)(((unsigned)tensor_k & 0xffffu) << 16);             // dim0[15:0]
    g1[2] = (int)((((unsigned)tensor_k >> 16) & 0xffffu) |
                  (((unsigned)tensor_rows & 0xffffu) << 16));        // dim0[31:16]|dim1[15:0]
    g1[3] = (int)((((unsigned)tensor_rows >> 16) & 0xffffu) |
                  (((unsigned)tile_k & 0xffffu) << 16));             // dim1[31:16]|tile_dim0
    g1[4] = (int)((unsigned)tile_rows & 0xffffu);                    // tile_dim1 (tile_dim2=0)
    g1[5] = (int)(unsigned)stride_k;                                 // dim0_stride[31:0]
    g1[6] = 0;                                                       // dim0_stride[47:32]
    g1[7] = 0;
    i32x4 z4 = {0, 0, 0, 0};
    i32x8 z8 = {0, 0, 0, 0, 0, 0, 0, 0};
    __builtin_amdgcn_tensor_load_to_lds(g0, g1, z4, z4, z8, 0);
}

// ---------------------------------------------------------------------------
// Elementwise f32 -> bf16 convert
// ---------------------------------------------------------------------------
__global__ void cvt_bf16_kernel(const float* __restrict__ x, bf16* __restrict__ y, int n) {
    int i = blockIdx.x * 256 + threadIdx.x;
    if (i < n) y[i] = (bf16)x[i];
}

// W[k][n] f32 row-major  ->  Wt[n][k] bf16 (column-contiguous B operand)
__global__ void cvt_w_transpose_kernel(const float* __restrict__ W, bf16* __restrict__ Wt) {
    int i = blockIdx.x * 256 + threadIdx.x;   // 0 .. D*D-1
    int n = i >> 10;
    int k = i & 1023;
    Wt[i] = (bf16)W[((size_t)k << 10) + n];
}

// ---------------------------------------------------------------------------
// TDM-fed bf16 GEMM: C[M,N] = A[M,K] * Bt[N,K]^T + bias
// Block = 256 threads (8 waves), block tile 128x64. Each wave: 16 rows x 64 cols
// (4 accumulators, A fragment reused x4). B panel (64 cols x 32 K) staged in LDS
// by the Tensor Data Mover, double buffered, waves read fragments via ds_load.
// out_mode 0: bf16 row-major; 1: bf16 transposed (outb[col*M+row]); 2: f32 + bias
// ---------------------------------------------------------------------------
__global__ void gemm_bf16_tdm_kernel(const bf16* __restrict__ A, const bf16* __restrict__ Bt,
                                     const float* __restrict__ bias,
                                     bf16* __restrict__ outb, float* __restrict__ outf,
                                     int M, int N, int K, int out_mode) {
    __shared__ bf16 shB[2][64 * 32];   // [buf][n_local*32 + k_local], 4 KB each
    const int lane = threadIdx.x & 31;
    const int wave = threadIdx.x >> 5;                 // 0..7
    const int tM = blockIdx.x * 128 + wave * 16;
    const int tN = blockIdx.y * 64;
    const int nchunks = K >> 5;

    if (wave == 0) {
        tdm_load_2d((unsigned)(size_t)&shB[0][0], Bt + (size_t)tN * K,
                    K, N, 32, 64, K);
        __builtin_amdgcn_s_wait_tensorcnt(0);
    }
    __syncthreads();

    v8f c[4] = {};
    for (int kc = 0; kc < nchunks; ++kc) {
        const int buf = kc & 1;
        if (wave == 0 && (kc + 1) < nchunks) {
            tdm_load_2d((unsigned)(size_t)&shB[buf ^ 1][0],
                        Bt + (size_t)tN * K + (size_t)(kc + 1) * 32,
                        K, N, 32, 64, K);
        }
        __builtin_prefetch(A + (size_t)(tM + (lane & 15)) * K + kc * 32 + 64, 0, 0);
        v16bf a = load_a_bf16(A, K, tM, kc * 32, lane);
        const bf16* shp = &shB[buf][0];
#pragma unroll
        for (int j = 0; j < 4; ++j) {
            v16bf b = load_b_bf16(shp, 32, j * 16, 0, lane);
            c[j] = wmma_bf16(a, b, c[j]);
        }
        if (wave == 0) __builtin_amdgcn_s_wait_tensorcnt(0);
        __syncthreads();
    }

    const int rowoff = (lane & 16) ? 8 : 0;
#pragma unroll
    for (int j = 0; j < 4; ++j) {
        const int col = tN + j * 16 + (lane & 15);
        const float bv = bias ? bias[col] : 0.0f;
#pragma unroll
        for (int r = 0; r < 8; ++r) {
            const int row = tM + r + rowoff;
            const float val = c[j][r] + bv;
            if (out_mode == 0)      outb[(size_t)row * N + col] = (bf16)val;
            else if (out_mode == 1) outb[(size_t)col * M + row] = (bf16)val;
            else                    outf[(size_t)row * N + col] = val;
        }
    }
}

// ---------------------------------------------------------------------------
// Scores: logits[b,h,s,t] = (qh . kh)/8 + causal mask.
// Each wave computes a 16x64 strip: A fragments (a0,a1) reused over 4 sub-tiles.
// Fully-masked sub-tiles skip WMMA and store -1e9.
// grid: x = (128 s-tiles * 32 t-quads)/4 = 1024, y = B*H
// ---------------------------------------------------------------------------
__global__ void attn_scores_kernel(const bf16* __restrict__ qb, const bf16* __restrict__ kb,
                                   float* __restrict__ attn) {
    const int lane = threadIdx.x & 31;
    const int wave = threadIdx.x >> 5;
    const int bh = blockIdx.y;
    const int b = bh >> 4, h = bh & 15;
    const int tile_id = blockIdx.x * 4 + wave;        // 0..4095
    const int tS = (tile_id >> 5) << 4;               // query tile (16 rows)
    const int tT0 = (tile_id & 31) << 6;              // key quad start (64 cols)

    float* out = attn + (size_t)bh * S_ * S_;
    const int rowoff = (lane & 16) ? 8 : 0;

    if (tT0 > tS) {   // entire 16x64 strip above the diagonal
#pragma unroll
        for (int j = 0; j < 4; ++j) {
            const int col = tT0 + j * 16 + (lane & 15);
#pragma unroll
            for (int r = 0; r < 8; ++r)
                out[(size_t)(tS + r + rowoff) * S_ + col] = NEGINF_;
        }
        return;
    }

    const bf16* Abase = qb + (size_t)b * S_ * D_ + h * DEPTH_;   // lda = D
    const bf16* Bbase = kb + (size_t)b * S_ * D_ + h * DEPTH_;   // row t contiguous in d
    const v16bf a0 = load_a_bf16(Abase, D_, tS, 0, lane);
    const v16bf a1 = load_a_bf16(Abase, D_, tS, 32, lane);
    const float scale = 0.125f;   // 1/sqrt(64)

#pragma unroll
    for (int j = 0; j < 4; ++j) {
        const int tT = tT0 + j * 16;
        const int col = tT + (lane & 15);
        if (tT > tS) {            // sub-tile fully masked
#pragma unroll
            for (int r = 0; r < 8; ++r)
                out[(size_t)(tS + r + rowoff) * S_ + col] = NEGINF_;
        } else {
            v8f c = {};
            c = wmma_bf16(a0, load_b_bf16(Bbase, D_, tT, 0, lane), c);
            c = wmma_bf16(a1, load_b_bf16(Bbase, D_, tT, 32, lane), c);
#pragma unroll
            for (int r = 0; r < 8; ++r) {
                const int row = tS + r + rowoff;
                float v = c[r] * scale;
                if (col > row) v += NEGINF_;
                out[(size_t)row * S_ + col] = v;
            }
        }
    }
}

// ---------------------------------------------------------------------------
// Row softmax over length-2048 rows, one 256-thread block per row.
// ---------------------------------------------------------------------------
__global__ void softmax_kernel(float* __restrict__ attn) {
    __shared__ float red[256];
    const int tid = threadIdx.x;
    float* p = attn + (size_t)blockIdx.x * S_;

    float v[8];
#pragma unroll
    for (int i = 0; i < 8; ++i) v[i] = p[tid + i * 256];

    float m = v[0];
#pragma unroll
    for (int i = 1; i < 8; ++i) m = fmaxf(m, v[i]);
    red[tid] = m;
    __syncthreads();
    for (int s = 128; s > 0; s >>= 1) {
        if (tid < s) red[tid] = fmaxf(red[tid], red[tid + s]);
        __syncthreads();
    }
    const float mx = red[0];
    __syncthreads();

    float sum = 0.0f;
#pragma unroll
    for (int i = 0; i < 8; ++i) { v[i] = __expf(v[i] - mx); sum += v[i]; }
    red[tid] = sum;
    __syncthreads();
    for (int s = 128; s > 0; s >>= 1) {
        if (tid < s) red[tid] += red[tid + s];
        __syncthreads();
    }
    const float inv = 1.0f / red[0];
#pragma unroll
    for (int i = 0; i < 8; ++i) p[tid + i * 256] = v[i] * inv;
}

// ---------------------------------------------------------------------------
// ctx[b,h,s,d] = sum_t attn[b,h,s,t] * vh[b,h,t,d]
// Each wave computes 16 rows x full 64-deep head: 4 accumulators, A reused x4.
// vT layout: vT[(h*64+d)*BS + b*S + t] (K-contiguous B operand).
// Causal truncation: K loop stops at (tM+47)&~31.
// grid: x = 128/4 = 32, y = B*H
// ---------------------------------------------------------------------------
__global__ void attn_ctx_kernel(const float* __restrict__ attn, const bf16* __restrict__ vT,
                                bf16* __restrict__ ctxb) {
    const int lane = threadIdx.x & 31;
    const int wave = threadIdx.x >> 5;
    const int bh = blockIdx.y;
    const int b = bh >> 4, h = bh & 15;
    const int tM = (blockIdx.x * 4 + wave) << 4;      // s tile, 0..2032

    const float* Abase = attn + (size_t)bh * S_ * S_;                    // lda = S
    const bf16*  Bbase = vT + (size_t)h * DEPTH_ * BS_ + (size_t)b * S_; // ldb = BS

    const int kmax = (tM + 47) & ~31;   // covers t <= tM+15, rounded up to 32
    v8f c[4] = {};
    for (int k0 = 0; k0 < kmax; k0 += 32) {
        v16bf a = load_a_f32(Abase, S_, tM, k0, lane);
#pragma unroll
        for (int j = 0; j < 4; ++j) {
            v16bf bfr = load_b_bf16(Bbase, BS_, j * 16, k0, lane);
            c[j] = wmma_bf16(a, bfr, c[j]);
        }
    }

    const int rowoff = (lane & 16) ? 8 : 0;
#pragma unroll
    for (int j = 0; j < 4; ++j) {
        const int col = j * 16 + (lane & 15);
#pragma unroll
        for (int r = 0; r < 8; ++r) {
            const int row = tM + r + rowoff;
            ctxb[((size_t)b * S_ + row) * D_ + h * DEPTH_ + col] = (bf16)c[j][r];
        }
    }
}

// ---------------------------------------------------------------------------
// Launch
// ---------------------------------------------------------------------------
extern "C" void kernel_launch(void* const* d_in, const int* in_sizes, int n_in,
                              void* d_out, int out_size, void* d_ws, size_t ws_size,
                              hipStream_t stream) {
    (void)in_sizes; (void)n_in; (void)out_size; (void)ws_size;
    // setup_inputs order: k, v, q, mask, wq_w, wq_b, wk_w, wk_b, wv_w, wv_b, wo_w, wo_b
    const float* in_k = (const float*)d_in[0];
    const float* in_v = (const float*)d_in[1];
    const float* in_q = (const float*)d_in[2];
    /* mask (d_in[3]) is the causal triu mask; applied analytically */
    const float* wq_w = (const float*)d_in[4];
    const float* wq_b = (const float*)d_in[5];
    const float* wk_w = (const float*)d_in[6];
    const float* wk_b = (const float*)d_in[7];
    const float* wv_w = (const float*)d_in[8];
    const float* wv_b = (const float*)d_in[9];
    const float* wo_w = (const float*)d_in[10];
    const float* wo_b = (const float*)d_in[11];

    float* out  = (float*)d_out;                       // [B,S,D]
    float* attn = out + (size_t)B_ * S_ * D_;          // [B,H,S,S]

    char* ws = (char*)d_ws;
    const size_t XB = (size_t)BS_ * D_ * sizeof(bf16); // 8 MiB
    const size_t WB = (size_t)D_ * D_ * sizeof(bf16);  // 2 MiB
    bf16* xkb  = (bf16*)(ws);
    bf16* xvb  = (bf16*)(ws + XB);
    bf16* xqb  = (bf16*)(ws + 2 * XB);
    bf16* Wtq  = (bf16*)(ws + 3 * XB);
    bf16* Wtk  = (bf16*)(ws + 3 * XB + WB);
    bf16* Wtv  = (bf16*)(ws + 3 * XB + 2 * WB);
    bf16* Wto  = (bf16*)(ws + 3 * XB + 3 * WB);
    bf16* qpb  = (bf16*)(ws + 3 * XB + 4 * WB);
    bf16* kpb  = (bf16*)(ws + 4 * XB + 4 * WB);
    bf16* vT   = (bf16*)(ws + 5 * XB + 4 * WB);
    bf16* ctxb = (bf16*)(ws + 6 * XB + 4 * WB);        // total 64 MiB

    const int NX = BS_ * D_;                           // 4M elements
    cvt_bf16_kernel<<<NX / 256, 256, 0, stream>>>(in_k, xkb, NX);
    cvt_bf16_kernel<<<NX / 256, 256, 0, stream>>>(in_v, xvb, NX);
    cvt_bf16_kernel<<<NX / 256, 256, 0, stream>>>(in_q, xqb, NX);

    const int NW = D_ * D_;                            // 1M elements
    cvt_w_transpose_kernel<<<NW / 256, 256, 0, stream>>>(wq_w, Wtq);
    cvt_w_transpose_kernel<<<NW / 256, 256, 0, stream>>>(wk_w, Wtk);
    cvt_w_transpose_kernel<<<NW / 256, 256, 0, stream>>>(wv_w, Wtv);
    cvt_w_transpose_kernel<<<NW / 256, 256, 0, stream>>>(wo_w, Wto);

    const dim3 ggrid(BS_ / 128, D_ / 64);              // 32 x 16 blocks
    gemm_bf16_tdm_kernel<<<ggrid, 256, 0, stream>>>(xqb, Wtq, wq_b, qpb, nullptr,
                                                    BS_, D_, D_, 0);
    gemm_bf16_tdm_kernel<<<ggrid, 256, 0, stream>>>(xkb, Wtk, wk_b, kpb, nullptr,
                                                    BS_, D_, D_, 0);
    gemm_bf16_tdm_kernel<<<ggrid, 256, 0, stream>>>(xvb, Wtv, wv_b, vT, nullptr,
                                                    BS_, D_, D_, 1);   // transposed store

    attn_scores_kernel<<<dim3(1024, B_ * H_), 128, 0, stream>>>(qpb, kpb, attn);
    softmax_kernel<<<B_ * H_ * S_, 256, 0, stream>>>(attn);
    attn_ctx_kernel<<<dim3(32, B_ * H_), 128, 0, stream>>>(attn, vT, ctxb);

    gemm_bf16_tdm_kernel<<<ggrid, 256, 0, stream>>>(ctxb, Wto, wo_b, nullptr, out,
                                                    BS_, D_, D_, 2);   // f32 out + bias
}
